// ImprovedGATDecoder_57028575756674
// MI455X (gfx1250) — compile-verified, matrix operand order checked by reading, not run on hardware
//
#include <hip/hip_runtime.h>

#define B_  1024
#define N_  101
#define H_  128
#define NH_ 8
#define HD_ 16

typedef float    v2f  __attribute__((ext_vector_type(2)));
typedef float    v8f  __attribute__((ext_vector_type(8)));
typedef _Float16 v16h __attribute__((ext_vector_type(16)));
typedef unsigned int u32x4 __attribute__((ext_vector_type(4)));
typedef int          i32x4 __attribute__((ext_vector_type(4)));
typedef int          i32x8 __attribute__((ext_vector_type(8)));

#define NEG_INF (-__builtin_inff())

// ---------------------------------------------------------------------------
// Kernel 1: encoder projections  O[m,j] = sum_k In[m,k] * W[k,j]
// One wave computes a 16x16 output tile with fp32 WMMA (16x16x4), K-loop of 32.
// ---------------------------------------------------------------------------
__global__ __launch_bounds__(256) void proj_gemm_k(
    const float* __restrict__ In,
    const float* __restrict__ W0, const float* __restrict__ W1,
    const float* __restrict__ W2,
    float* __restrict__ O0, float* __restrict__ O1, float* __restrict__ O2)
{
  const float* W = (blockIdx.y == 0) ? W0 : (blockIdx.y == 1) ? W1 : W2;
  float*       O = (blockIdx.y == 0) ? O0 : (blockIdx.y == 1) ? O1 : O2;

  const int wave = threadIdx.x >> 5;
  const int lane = threadIdx.x & 31;
  const int lm   = lane & 15;
  const int lh   = lane >> 4;
  const int m0   = blockIdx.x * 16;
  const int c0   = wave * 16;

  const float* arow = In + (size_t)(m0 + lm) * H_;
  v8f acc = {};

#if __has_builtin(__builtin_amdgcn_wmma_f32_16x16x4_f32)
  for (int k0 = 0; k0 < H_; k0 += 4) {
    const int kb = k0 + 2 * lh;
    v2f a; a.x = arow[kb];                    a.y = arow[kb + 1];
    v2f b; b.x = W[kb * H_ + c0 + lm];        b.y = W[(kb + 1) * H_ + c0 + lm];
    acc = __builtin_amdgcn_wmma_f32_16x16x4_f32(false, a, false, b,
                                                (short)0, acc, false, false);
  }
#else
  for (int k0 = 0; k0 < H_; k0 += 32) {
    v16h a, b;
#pragma unroll
    for (int v = 0; v < 8; ++v) {
      int ka = k0 + ((v < 4) ? (2 * v + 8 * lh) : (16 + 2 * (v - 4) + 8 * lh));
      a[2 * v]     = (_Float16)arow[ka];
      a[2 * v + 1] = (_Float16)arow[ka + 1];
      b[2 * v]     = (_Float16)W[ka * H_ + c0 + lm];
      b[2 * v + 1] = (_Float16)W[(ka + 1) * H_ + c0 + lm];
    }
    acc = __builtin_amdgcn_wmma_f32_16x16x32_f16(false, a, false, b,
                                                 (short)0, acc, false, false);
  }
#endif

#pragma unroll
  for (int v = 0; v < 8; ++v)
    O[(size_t)(m0 + v + 8 * lh) * H_ + c0 + lm] = acc[v];
}

// ---------------------------------------------------------------------------
// Block reduction helper (blockDim.x == 128)
// ---------------------------------------------------------------------------
__device__ __forceinline__ float block_sum(float v, float* sRed, int tid) {
  sRed[tid] = v;
  __syncthreads();
#pragma unroll
  for (int s = 64; s > 0; s >>= 1) {
    if (tid < s) sRed[tid] += sRed[tid + s];
    __syncthreads();
  }
  float r = sRed[0];
  __syncthreads();
  return r;
}

// ---------------------------------------------------------------------------
// TDM: 1-D tensor DMA, global -> LDS, data_size = 8 bytes.
// nelem8 = number of 8-byte elements. lds_off = LDS byte address.
// ---------------------------------------------------------------------------
#if __has_builtin(__builtin_amdgcn_tensor_load_to_lds) && \
    __has_builtin(__builtin_amdgcn_s_wait_tensorcnt)
#define HAVE_TDM 1
__device__ __forceinline__ void tdm_load_1d(const void* gsrc, unsigned lds_off,
                                            unsigned nelem8) {
  const unsigned long long ga = (unsigned long long)(size_t)gsrc;
  u32x4 g0;
  g0.x = 1u;                                  // count=1 (valid user descriptor)
  g0.y = lds_off;                             // lds_addr  (bits 63:32)
  g0.z = (unsigned)(ga & 0xffffffffu);        // global_addr[31:0]
  g0.w = (unsigned)((ga >> 32) & 0x01ffffffu) // global_addr[56:32]
         | 0x80000000u;                       // type=2 ("image")
  i32x8 g1;
  g1[0] = (int)(3u << 16);                    // data_size=3 (8B)
  g1[1] = (int)((nelem8 & 0xffffu) << 16);    // tensor_dim0[15:0]  @63:48
  g1[2] = (int)(((nelem8 >> 16) & 0xffffu)    // tensor_dim0[31:16] @79:64
                | (1u << 16));                // tensor_dim1=1      @111:80 (lo)
  g1[3] = (int)((nelem8 & 0xffffu) << 16);    // tile_dim0 @127:112 (dim1 hi=0)
  g1[4] = (int)1;                             // tile_dim1=1, tile_dim2=0
  g1[5] = (int)nelem8;                        // tensor_dim0_stride[31:0]
  g1[6] = 0;                                  // stride hi / dim1_stride lo
  g1[7] = 0;
  i32x4 z4 = {0, 0, 0, 0};
#if defined(__clang_major__) && (__clang_major__ >= 23)
  i32x8 z8 = {0, 0, 0, 0, 0, 0, 0, 0};
  __builtin_amdgcn_tensor_load_to_lds(g0, g1, z4, z4, z8, 0);
#else
  __builtin_amdgcn_tensor_load_to_lds(g0, g1, z4, z4, 0);
#endif
}
#endif

// ---------------------------------------------------------------------------
// Kernel 2: full decode, one WG (128 thr) per batch row.
// K/V/Kp live in LDS (TDM-preloaded); wq/wo/fc_w columns live in VGPRs.
// ---------------------------------------------------------------------------
__global__ void __launch_bounds__(128, 1) decode_k(
    const float* __restrict__ E,          // (B,N,H)
    const float* __restrict__ pool_proc,  // (B,H)   ws
    const float* __restrict__ capacity,   // (B,1)
    const float* __restrict__ demand,     // (B,N)
    const float* __restrict__ fc_w,       // (H+1,H)
    const float* __restrict__ dec_g, const float* __restrict__ dec_b,
    const float* __restrict__ wq,   const float* __restrict__ wo,
    const float* __restrict__ ptr_g, const float* __restrict__ ptr_b,
    const float* __restrict__ Kh, const float* __restrict__ Vh,
    const float* __restrict__ Kp,         // (B,N,H) ws each
    const int* __restrict__ n_steps_p, const int* __restrict__ T_p,
    float* __restrict__ out)              // actions (B*(S+2)) then log_p (B)
{
  extern __shared__ float smem[];
  const int b   = blockIdx.x;
  const int tid = threadIdx.x;

  float* sK    = smem;                  // N*H
  float* sV    = sK   + N_ * H_;        // N*H
  float* sP    = sV   + N_ * H_;        // N*H
  float* sInp  = sP   + N_ * H_;        // H
  float* sD    = sInp + H_;             // H
  float* sQ    = sD   + H_;             // H
  float* sX    = sQ   + H_;             // H
  float* sRed  = sX   + H_;             // 128
  float* sLog  = sRed + 128;            // NH*N
  float* sComp = sLog + NH_ * N_;       // N
  float* sMask = sComp + N_;            // N
  float* sMB   = sMask + N_;            // N
  float* sScal = sMB  + N_;             // [0]=cap [1]=lpsum
  int*   sInt  = (int*)(sScal + 4);     // [0]=idx
  int*   sIRed = (int*)(sScal + 8);     // 128 ints (argmax)

  // ---- preload per-batch K/V/Kp into LDS ----
#ifdef HAVE_TDM
  if (tid < 32) {   // one wave issues the DMAs
    const unsigned base = __builtin_amdgcn_groupstaticsize();
    const unsigned mbytes = (unsigned)(N_ * H_ * sizeof(float));
    const unsigned ne8 = (unsigned)(N_ * H_ / 2);
    tdm_load_1d(Kh + (size_t)b * N_ * H_, base + 0u * mbytes, ne8);
    tdm_load_1d(Vh + (size_t)b * N_ * H_, base + 1u * mbytes, ne8);
    tdm_load_1d(Kp + (size_t)b * N_ * H_, base + 2u * mbytes, ne8);
    __builtin_amdgcn_s_wait_tensorcnt(0);
  }
#else
  {
    const float4* gK = (const float4*)(Kh + (size_t)b * N_ * H_);
    const float4* gV = (const float4*)(Vh + (size_t)b * N_ * H_);
    const float4* gP = (const float4*)(Kp + (size_t)b * N_ * H_);
    float4* lK = (float4*)sK; float4* lV = (float4*)sV; float4* lP = (float4*)sP;
    for (int i = tid; i < N_ * H_ / 4; i += 128) {
      lK[i] = gK[i]; lV[i] = gV[i]; lP[i] = gP[i];
    }
  }
#endif

  // ---- per-thread register cache of weight columns (column tid) ----
  float fcw_col[H_], wq_col[H_], wo_col[H_];
#pragma unroll
  for (int k = 0; k < H_; ++k) {
    fcw_col[k] = fc_w[k * H_ + tid];
    wq_col[k]  = wq[k * H_ + tid];
    wo_col[k]  = wo[k * H_ + tid];
  }
  const float fcw_cap = fc_w[H_ * H_ + tid];
  const float pp_t    = pool_proc[(size_t)b * H_ + tid];
  const float dg = dec_g[tid], db = dec_b[tid];
  const float pg = ptr_g[tid], pb = ptr_b[tid];

  if (tid < H_) sInp[tid] = E[((size_t)b * N_) * H_ + tid];   // node 0
  for (int i = tid; i < N_; i += 128) sMask[i] = 0.f;
  if (tid == 0) { sScal[0] = capacity[b]; sScal[1] = 0.f; sInt[0] = 0; }
  __syncthreads();

  const int   S    = *n_steps_p;
  const float Tinv = 1.0f / (float)(*T_p);

  for (int i = 0; i < S; ++i) {
    const int   idx = sInt[0];
    const float cap = sScal[0];

    // mask1 update (uses carried idx)
    if (i > 0 && tid == idx && idx != 0 && tid < N_) sMask[tid] = 1.f;
    __syncthreads();

    // infeasibility of customers 1..N-1
    float infeas = 0.f;
    if (tid >= 1 && tid < N_)
      infeas = (sMask[tid] > 0.f || demand[(size_t)b * N_ + tid] > cap) ? 1.f : 0.f;
    const float cnt = block_sum((tid >= 1 && tid < N_) ? infeas : 0.f, sRed, tid);
    const bool any_feas = cnt < (float)(N_ - 1) - 0.5f;
    const bool all_cust = cnt > (float)(N_ - 1) - 0.5f;

    const float msum = block_sum((tid >= 1 && tid < N_) ? sMask[tid] : 0.f, sRed, tid);
    const bool  done = msum >= (float)(N_ - 1) - 0.5f;

    if (tid < N_) {
      if (tid == 0) {
        bool depot = (i == 0) ? true : (idx == 0 && any_feas);
        if (all_cust) depot = false;
        sMB[0] = depot ? 1.f : 0.f;
      } else {
        sMB[tid] = infeas;
      }
    }
    __syncthreads();

    // d = [inp,cap] @ fc_w + pool_proc, then LayerNorm
    {
      float acc = cap * fcw_cap + pp_t;
#pragma unroll
      for (int k = 0; k < H_; ++k) acc += sInp[k] * fcw_col[k];
      sD[tid] = acc;
    }
    float mu  = block_sum(sD[tid], sRed, tid) * (1.0f / H_);
    float xc  = sD[tid] - mu;
    float var = block_sum(xc * xc, sRed, tid) * (1.0f / H_);
    sD[tid] = xc * rsqrtf(var + 1e-5f) * dg + db;
    __syncthreads();

    // Qh = d @ wq
    {
      float acc = 0.f;
#pragma unroll
      for (int k = 0; k < H_; ++k) acc += sD[k] * wq_col[k];
      sQ[tid] = acc;
    }
    __syncthreads();

    // logits[h,n] = (Qh[h,:] . Kh[n,h,:]) / sqrt(HD), masked
    for (int t = tid; t < NH_ * N_; t += 128) {
      const int h = t / N_, n = t - h * N_;
      const float* kr = sK + n * H_ + h * HD_;
      const float* qr = sQ + h * HD_;
      float acc = 0.f;
#pragma unroll
      for (int d2 = 0; d2 < HD_; ++d2) acc += qr[d2] * kr[d2];
      sLog[t] = (sMB[n] > 0.f) ? NEG_INF : acc * 0.25f;
    }
    __syncthreads();

    // per-head softmax over n
    if (tid < NH_) {
      float m = NEG_INF;
      for (int n = 0; n < N_; ++n) m = fmaxf(m, sLog[tid * N_ + n]);
      if (m == NEG_INF) {
        for (int n = 0; n < N_; ++n) sLog[tid * N_ + n] = 0.f;
      } else {
        float s = 0.f;
        for (int n = 0; n < N_; ++n) {
          float e = __expf(sLog[tid * N_ + n] - m);
          sLog[tid * N_ + n] = e; s += e;
        }
        const float inv = 1.0f / s;
        for (int n = 0; n < N_; ++n) sLog[tid * N_ + n] *= inv;
      }
    }
    __syncthreads();

    // x = attn @ Vh
    {
      const int h = tid >> 4;
      float acc = 0.f;
      for (int n = 0; n < N_; ++n) acc += sLog[h * N_ + n] * sV[n * H_ + tid];
      sX[tid] = acc;
    }
    __syncthreads();

    // xo = x @ wo, then LayerNorm -> sQ (xn)
    {
      float acc = 0.f;
#pragma unroll
      for (int k = 0; k < H_; ++k) acc += sX[k] * wo_col[k];
      sD[tid] = acc;
    }
    mu  = block_sum(sD[tid], sRed, tid) * (1.0f / H_);
    xc  = sD[tid] - mu;
    var = block_sum(xc * xc, sRed, tid) * (1.0f / H_);
    sQ[tid] = xc * rsqrtf(var + 1e-5f) * pg + pb;
    __syncthreads();

    // comp[n] = (xn . Kp[n]) / sqrt(H), masked; 0 everywhere if all masked
    const bool allmb = (sMB[0] > 0.f) && all_cust;
    if (tid < N_) {
      const float* kr = sP + tid * H_;
      float acc = 0.f;
      for (int k = 0; k < H_; ++k) acc += sQ[k] * kr[k];
      float c = acc * 0.08838834764831845f;   // 1/sqrt(128)
      c = (sMB[tid] > 0.f) ? NEG_INF : c;
      if (allmb) c = 0.f;
      sComp[tid] = c;
    }
    __syncthreads();

    // max + first-argmax over comp/T
    const float v = (tid < N_) ? sComp[tid] * Tinv : NEG_INF;
    sRed[tid]  = v;
    sIRed[tid] = (tid < N_) ? tid : 0x7fffffff;
    __syncthreads();
#pragma unroll
    for (int s = 64; s > 0; s >>= 1) {
      if (tid < s) {
        const float vo = sRed[tid + s]; const int io = sIRed[tid + s];
        if (vo > sRed[tid] || (vo == sRed[tid] && io < sIRed[tid])) {
          sRed[tid] = vo; sIRed[tid] = io;
        }
      }
      __syncthreads();
    }
    const float M    = sRed[0];
    const int   amax = sIRed[0];
    __syncthreads();

    const float e    = (tid < N_) ? __expf(v - M) : 0.f;
    const float Ssum = block_sum(e, sRed, tid);

    if (tid == 0) {
      const float pa = (1.0f / Ssum + 1e-10f) / (1.0f + (float)N_ * 1e-10f);
      const float lp = done ? 0.f : logf(pa);
      sScal[1] += lp;
      sInt[0] = amax;
      const float sel = demand[(size_t)b * N_ + amax];
      sScal[0] = (amax == 0) ? capacity[b] : (cap - sel);
      out[(size_t)b * (S + 2) + 1 + i] = (float)amax;
    }
    __syncthreads();
    if (tid < H_) sInp[tid] = E[((size_t)b * N_ + sInt[0]) * H_ + tid];
    __syncthreads();
  }

  if (tid == 0) {
    out[(size_t)b * (S + 2)] = 0.f;
    const int last = sInt[0];
    out[(size_t)b * (S + 2) + S + 1] = (last == 0) ? -1.f : 0.f;
    out[(size_t)B_ * (S + 2) + b] = sScal[1];   // log_p
  }
}

// ---------------------------------------------------------------------------
extern "C" void kernel_launch(void* const* d_in, const int* in_sizes, int n_in,
                              void* d_out, int out_size, void* d_ws, size_t ws_size,
                              hipStream_t stream) {
  const float* E        = (const float*)d_in[0];
  const float* pool     = (const float*)d_in[1];
  const float* capacity = (const float*)d_in[2];
  const float* demand   = (const float*)d_in[3];
  const float* fc_w     = (const float*)d_in[4];
  const float* fc1_w    = (const float*)d_in[5];
  const float* dec_g    = (const float*)d_in[6];
  const float* dec_b    = (const float*)d_in[7];
  const float* wq       = (const float*)d_in[8];
  const float* wk       = (const float*)d_in[9];
  const float* wv       = (const float*)d_in[10];
  const float* wo       = (const float*)d_in[11];
  const float* kp_w     = (const float*)d_in[12];
  const float* ptr_g    = (const float*)d_in[13];
  const float* ptr_b    = (const float*)d_in[14];
  const int*   n_steps  = (const int*)d_in[15];
  const int*   T        = (const int*)d_in[16];

  float* ws = (float*)d_ws;
  const size_t KN = (size_t)B_ * N_ * H_;
  float* Kh = ws;
  float* Vh = ws + KN;
  float* Kp = ws + 2 * KN;
  float* pp = ws + 3 * KN;

  // Encoder projections: K, V, Kp (WMMA fp32)
  proj_gemm_k<<<dim3((B_ * N_) / 16, 3), 256, 0, stream>>>(
      E, wk, wv, kp_w, Kh, Vh, Kp);
  // pool_proc = pool @ fc1_w (WMMA fp32)
  proj_gemm_k<<<dim3(B_ / 16, 1), 256, 0, stream>>>(
      pool, fc1_w, fc1_w, fc1_w, pp, pp, pp);

  // Decode: one WG per batch row; K/V/Kp resident in LDS (TDM preload)
  const size_t shmem =
      (size_t)(3 * N_ * H_ + 5 * H_ + NH_ * N_ + 3 * N_ + 8) * sizeof(float)
      + 128 * sizeof(int) + 64;
  decode_k<<<dim3(B_), 128, shmem, stream>>>(
      E, pp, capacity, demand, fc_w, dec_g, dec_b, wq, wo, ptr_g, ptr_b,
      Kh, Vh, Kp, n_steps, T, (float*)d_out);
}
